// DisentangledStateEncoder_29291676959177
// MI455X (gfx1250) — matrix operand order repaired and checked by reading, compiled, not run on hardware
//
#include <hip/hip_runtime.h>
#include <hip/hip_bf16.h>
#include <math.h>

// ---------------------------------------------------------------------------
// DisentangledStateEncoder for MI455X (gfx1250, wave32)
//   Kernel P: prepack W_in / W_c1[5:] / W_innov into bf16 WMMA B-fragments
//   Kernel A: h = gelu(LN(x@W_in+b_in)); hpart = h@W_c1[5:]+b_c1;
//             bx = h@W_innov+b_innov   (bf16 WMMA, f32 accum, weights in LDS)
//   Kernel B: 2048-step nonlinear recurrence, one wave per batch row
// ---------------------------------------------------------------------------

typedef __attribute__((ext_vector_type(16))) __bf16 v16bf;
typedef __attribute__((ext_vector_type(8)))  float  v8f;

#define B_   256
#define S_   2048
#define DIN  128
#define H_   64
#define SD_  5
#define LN_EPS 1e-5f

// fragment table: f 0..15  -> W_in   (kb = f>>2, nt = f&3)
//                 f 16..23 -> W_c1s  (kb = (f-16)>>2, nt = (f-16)&3)
//                 f 24..25 -> W_innov (kb = f-24, nt = 0, cols 5..15 zero)
#define NFRAG 26
#define WPACK_ELEMS (NFRAG * 32 * 16)   // bf16 elements = 26624 bytes

__device__ __forceinline__ float gelu_exact(float x) {
    return 0.5f * x * (1.0f + erff(x * 0.70710678118654752f));
}
__device__ __forceinline__ float sigmoidf_(float x) {
    return 1.0f / (1.0f + expf(-x));
}

// ---------------------------------------------------------------------------
// Kernel P: single block. B-fragment layout per the CDNA5 ISA:
//   lane L: n = (L&15) + nt*16 ; element i: k = kb*32 + (L>=16?8:0) + (i&7) + (i>=8?16:0)
// Stored contiguously: wpack[(f*32 + L)*16 + i]
// ---------------------------------------------------------------------------
__global__ __launch_bounds__(256) void pack_weights(
    const float* __restrict__ W_in, const float* __restrict__ W_c1,
    const float* __restrict__ W_innov, __bf16* __restrict__ wpack)
{
    for (int slot = threadIdx.x; slot < NFRAG * 32; slot += 256) {
        const int f     = slot >> 5;
        const int L     = slot & 31;
        const int n16   = L & 15;
        const int khalf = (L >> 4) * 8;
        __bf16 vals[16];
        if (f < 16) {
            const int kb = f >> 2, nt = f & 3;
            const int n = n16 + nt * 16;
#pragma unroll
            for (int i = 0; i < 16; ++i)
                vals[i] = (__bf16)W_in[(kb * 32 + khalf + (i & 7) + ((i >> 3) << 4)) * H_ + n];
        } else if (f < 24) {
            const int kb = (f - 16) >> 2, nt = (f - 16) & 3;
            const int n = n16 + nt * 16;
#pragma unroll
            for (int i = 0; i < 16; ++i)
                vals[i] = (__bf16)W_c1[(SD_ + kb * 32 + khalf + (i & 7) + ((i >> 3) << 4)) * H_ + n];
        } else {
            const int kb = f - 24;
            const int mm = (n16 < SD_) ? n16 : 0;       // keep loads in-bounds
#pragma unroll
            for (int i = 0; i < 16; ++i) {
                const float wv = W_innov[(kb * 32 + khalf + (i & 7) + ((i >> 3) << 4)) * SD_ + mm];
                vals[i] = (n16 < SD_) ? (__bf16)wv : (__bf16)0.0f;
            }
        }
#pragma unroll
        for (int i = 0; i < 16; ++i) wpack[slot * 16 + i] = vals[i];
    }
}

// ---------------------------------------------------------------------------
// Kernel A: one wave per 16-row tile of the (B*S, DIN) activation matrix.
// 256 threads/block = 8 waves = 8 tiles per block. Weights live in LDS.
// ---------------------------------------------------------------------------
__global__ __launch_bounds__(256) void encoder_front(
    const float* __restrict__ x,      const __bf16* __restrict__ wpack,
    const float* __restrict__ b_in,   const float* __restrict__ ln_g,
    const float* __restrict__ ln_b,   const float* __restrict__ b_innov,
    const float* __restrict__ b_c1,
    float* __restrict__ hpart, float* __restrict__ bx)
{
    __shared__ __align__(32) __bf16 ldsw[WPACK_ELEMS];   // 26 KB packed weights
    __shared__ __bf16 hstage[8][16][72];                 // per-wave h tile

    // cooperative global->LDS weight copy (coalesced b128)
    {
        const uint4* src = (const uint4*)wpack;
        uint4*       dst = (uint4*)ldsw;
        for (int i = threadIdx.x; i < (WPACK_ELEMS * 2) / 16; i += 256)
            dst[i] = src[i];
    }
    __syncthreads();

    const int  lane = threadIdx.x & 31;
    const int  w    = threadIdx.x >> 5;
    const long tile = (long)blockIdx.x * 8 + w;     // 16-row tile index
    const long rowBase = tile * 16;
    const int  m  = lane & 15;                      // A-row / C-col index
    const int  hi = lane >> 4;                      // half-wave selector

    // B-fragment fetch from LDS: 32 contiguous bytes per lane (2x ds_load_b128)
    const __bf16* lw = ldsw + lane * 16;
#define LOADFRAG(f) (*(const v16bf*)(lw + (f) * 32 * 16))

    // ---- A fragments: x rows as bf16, 4 chunks of K=32 ------------------
    v16bf afrag[4];
#pragma unroll
    for (int kb = 0; kb < 4; ++kb) {
        const float* xr = x + (rowBase + m) * DIN + kb * 32 + hi * 8;
#pragma unroll
        for (int i = 0; i < 8; ++i) {
            afrag[kb][i]     = (__bf16)xr[i];
            afrag[kb][i + 8] = (__bf16)xr[16 + i];
        }
    }

    // ---- h = x @ W_in  (4 N-tiles of 16 cols, K=128 in 4 WMMAs each) ----
    v8f acc[4];
#pragma unroll
    for (int nt = 0; nt < 4; ++nt) {
        v8f c = {};
#pragma unroll
        for (int kb = 0; kb < 4; ++kb) {
            v16bf bfrag = LOADFRAG(kb * 4 + nt);
            c = __builtin_amdgcn_wmma_f32_16x16x32_bf16(
                    false, afrag[kb], false, bfrag, (short)0, c, false, false);
        }
        acc[nt] = c;
    }

    // ---- +b_in, LayerNorm over the 64 cols of each row, gelu ------------
    // C layout: acc[nt][v] is (row = v + hi*8, col = m + nt*16)
#pragma unroll
    for (int nt = 0; nt < 4; ++nt) {
        const float bn = b_in[m + nt * 16];
#pragma unroll
        for (int v = 0; v < 8; ++v) acc[nt][v] += bn;
    }
    v8f rsum = acc[0] + acc[1] + acc[2] + acc[3];
    v8f rsq  = acc[0]*acc[0] + acc[1]*acc[1] + acc[2]*acc[2] + acc[3]*acc[3];
#pragma unroll
    for (int mask = 8; mask >= 1; mask >>= 1) {   // reduce over 16-lane half
#pragma unroll
        for (int v = 0; v < 8; ++v) {
            rsum[v] += __shfl_xor(rsum[v], mask, 32);
            rsq[v]  += __shfl_xor(rsq[v],  mask, 32);
        }
    }
#pragma unroll
    for (int nt = 0; nt < 4; ++nt) {
        const float g  = ln_g[m + nt * 16];
        const float bb = ln_b[m + nt * 16];
#pragma unroll
        for (int v = 0; v < 8; ++v) {
            const float mu  = rsum[v] * (1.0f / 64.0f);
            const float var = rsq[v] * (1.0f / 64.0f) - mu * mu;
            float hn = (acc[nt][v] - mu) * rsqrtf(var + LN_EPS) * g + bb;
            hn = gelu_exact(hn);
            hstage[w][v + hi * 8][m + nt * 16] = (__bf16)hn;
        }
    }

    // ---- re-fragment h (C layout -> A layout) via LDS; K=64 -> 2 chunks -
    v16bf ha[2];
#pragma unroll
    for (int kb = 0; kb < 2; ++kb) {
#pragma unroll
        for (int i = 0; i < 8; ++i) {
            ha[kb][i]     = hstage[w][m][kb * 32 + hi * 8 + i];
            ha[kb][i + 8] = hstage[w][m][kb * 32 + 16 + hi * 8 + i];
        }
    }

    // ---- hpart = h @ W_c1[SD:] + b_c1  (64x64, 8 WMMAs) -----------------
#pragma unroll
    for (int nt = 0; nt < 4; ++nt) {
        v8f c = {};
#pragma unroll
        for (int kb = 0; kb < 2; ++kb) {
            v16bf bfrag = LOADFRAG(16 + kb * 4 + nt);
            c = __builtin_amdgcn_wmma_f32_16x16x32_bf16(
                    false, ha[kb], false, bfrag, (short)0, c, false, false);
        }
        const float bc = b_c1[m + nt * 16];
#pragma unroll
        for (int v = 0; v < 8; ++v)
            hpart[(rowBase + v + hi * 8) * H_ + (m + nt * 16)] = c[v] + bc;
    }

    // ---- bx = h @ W_innov + b_innov  (N=5 in a zero-padded 16-col tile) -
    {
        v8f c = {};
#pragma unroll
        for (int kb = 0; kb < 2; ++kb) {
            v16bf bfrag = LOADFRAG(24 + kb);
            c = __builtin_amdgcn_wmma_f32_16x16x32_bf16(
                    false, ha[kb], false, bfrag, (short)0, c, false, false);
        }
        if (m < SD_) {
            const float bi = b_innov[m];
#pragma unroll
            for (int v = 0; v < 8; ++v)
                bx[(rowBase + v + hi * 8) * SD_ + m] = c[v] + bi;
        }
    }
#undef LOADFRAG
}

// ---------------------------------------------------------------------------
// Kernel B: one wave per batch row. Lanes cover the 64-dim of u (2 cols each);
// the 5-dim correction uses 5 wave32 butterfly reductions per step.
// ---------------------------------------------------------------------------
__global__ __launch_bounds__(256) void encoder_scan(
    const float* __restrict__ W_c1, const float* __restrict__ W_c2,
    const float* __restrict__ b_c2, const float* __restrict__ corr_scale,
    const float* __restrict__ raw_alpha_L, const float* __restrict__ raw_alpha_T,
    const float* __restrict__ raw_gamma,   const float* __restrict__ raw_alpha_R,
    const float* __restrict__ omega,
    const float* __restrict__ hpart, const float* __restrict__ bx,
    float* __restrict__ out)
{
    const int lane = threadIdx.x & 31;
    const int b    = blockIdx.x * 8 + (threadIdx.x >> 5);

    const float aL = sigmoidf_(raw_alpha_L[0]) * 0.15f + 0.85f;
    const float aT = sigmoidf_(raw_alpha_T[0]) * 0.25f + 0.70f;
    const float g  = sigmoidf_(raw_gamma[0])   * 0.20f + 0.80f;
    const float aR = sigmoidf_(raw_alpha_R[0]) * 0.40f;
    const float om = omega[0];
    const float r00 =  g * cosf(om), r01 = -g * sinf(om);
    const float r10 =  g * sinf(om), r11 =  g * cosf(om);
    const float cs = corr_scale[0];

    // Per-lane weight slices: this lane owns columns j0=lane, j1=lane+32.
    float wc1a[SD_], wc1b[SD_], wc2a[SD_], wc2b[SD_], bc2[SD_];
#pragma unroll
    for (int k = 0; k < SD_; ++k) {
        wc1a[k] = W_c1[k * H_ + lane];          // Wc1_s = W_c1[:SD]
        wc1b[k] = W_c1[k * H_ + lane + 32];
        wc2a[k] = W_c2[lane * SD_ + k];
        wc2b[k] = W_c2[(lane + 32) * SD_ + k];
        bc2[k]  = b_c2[k];
    }

    float s0 = 0.f, s1 = 0.f, s2 = 0.f, s3 = 0.f, s4 = 0.f;
    const float* hp  = hpart + (long)b * S_ * H_;
    const float* bxp = bx    + (long)b * S_ * SD_;

    for (int t = 0; t < S_; ++t) {
        __builtin_prefetch(hp + (t + 8) * H_ + lane, 0, 1);
        const float h0 = hp[t * H_ + lane];
        const float h1 = hp[t * H_ + lane + 32];
        const float b0v = bxp[t * SD_ + 0];
        const float b1v = bxp[t * SD_ + 1];
        const float b2v = bxp[t * SD_ + 2];
        const float b3v = bxp[t * SD_ + 3];
        const float b4v = bxp[t * SD_ + 4];

        // s_lin = diag/rot recurrence applied to s, plus bx_t (uniform per wave)
        const float l0 = s0 * aL + b0v;
        const float l1 = s1 * aT + b1v;
        const float l2 = s2 * r00 + s3 * r10 + b2v;
        const float l3 = s2 * r01 + s3 * r11 + b3v;
        const float l4 = s4 * aR + b4v;

        // u = gelu(s_lin @ Wc1_s + hp_t): 2 columns per lane
        float u0 = h0 + l0*wc1a[0] + l1*wc1a[1] + l2*wc1a[2] + l3*wc1a[3] + l4*wc1a[4];
        float u1 = h1 + l0*wc1b[0] + l1*wc1b[1] + l2*wc1b[2] + l3*wc1b[3] + l4*wc1b[4];
        u0 = gelu_exact(u0);
        u1 = gelu_exact(u1);

        // corr = cs * tanh(u @ W_c2 + b_c2): 5 wave reductions
        float p0 = u0*wc2a[0] + u1*wc2b[0];
        float p1 = u0*wc2a[1] + u1*wc2b[1];
        float p2 = u0*wc2a[2] + u1*wc2b[2];
        float p3 = u0*wc2a[3] + u1*wc2b[3];
        float p4 = u0*wc2a[4] + u1*wc2b[4];
#pragma unroll
        for (int mask = 16; mask >= 1; mask >>= 1) {
            p0 += __shfl_xor(p0, mask, 32);
            p1 += __shfl_xor(p1, mask, 32);
            p2 += __shfl_xor(p2, mask, 32);
            p3 += __shfl_xor(p3, mask, 32);
            p4 += __shfl_xor(p4, mask, 32);
        }
        s0 = l0 + cs * tanhf(p0 + bc2[0]);
        s1 = l1 + cs * tanhf(p1 + bc2[1]);
        s2 = l2 + cs * tanhf(p2 + bc2[2]);
        s3 = l3 + cs * tanhf(p3 + bc2[3]);
        s4 = l4 + cs * tanhf(p4 + bc2[4]);
    }

    if (lane == 0) {
        out[b * SD_ + 0] = s0;
        out[b * SD_ + 1] = s1;
        out[b * SD_ + 2] = s2;
        out[b * SD_ + 3] = s3;
        out[b * SD_ + 4] = s4;
    }
}

// ---------------------------------------------------------------------------
extern "C" void kernel_launch(void* const* d_in, const int* in_sizes, int n_in,
                              void* d_out, int out_size, void* d_ws, size_t ws_size,
                              hipStream_t stream)
{
    const float* x           = (const float*)d_in[0];
    const float* W_in        = (const float*)d_in[1];
    const float* b_in        = (const float*)d_in[2];
    const float* ln_g        = (const float*)d_in[3];
    const float* ln_b        = (const float*)d_in[4];
    const float* W_innov     = (const float*)d_in[5];
    const float* b_innov     = (const float*)d_in[6];
    const float* W_c1        = (const float*)d_in[7];
    const float* b_c1        = (const float*)d_in[8];
    const float* W_c2        = (const float*)d_in[9];
    const float* b_c2        = (const float*)d_in[10];
    const float* corr_scale  = (const float*)d_in[11];
    const float* raw_alpha_L = (const float*)d_in[12];
    const float* raw_alpha_T = (const float*)d_in[13];
    const float* raw_gamma   = (const float*)d_in[14];
    const float* raw_alpha_R = (const float*)d_in[15];
    const float* omega       = (const float*)d_in[16];

    const size_t HB = (size_t)B_ * S_ * H_  * sizeof(float);   // 134217728
    const size_t BB = (size_t)B_ * S_ * SD_ * sizeof(float);   // 10485760
    float*  hpart = (float*)d_ws;
    float*  bx    = (float*)((char*)d_ws + HB);
    __bf16* wpack = (__bf16*)((char*)d_ws + HB + BB);

    // Kernel P: pack weights into WMMA B-fragment layout (one block)
    hipLaunchKernelGGL(pack_weights, dim3(1), dim3(256), 0, stream,
                       W_in, W_c1, W_innov, wpack);

    // Kernel A: 524288 rows / 16 rows-per-wave / 8 waves-per-block = 4096 blocks
    hipLaunchKernelGGL(encoder_front, dim3((B_ * S_ / 16) / 8), dim3(256), 0, stream,
                       x, wpack, b_in, ln_g, ln_b, b_innov, b_c1,
                       hpart, bx);

    // Kernel B: 256 batch rows, one wave each, 8 waves/block = 32 blocks
    hipLaunchKernelGGL(encoder_scan, dim3(B_ / 8), dim3(256), 0, stream,
                       W_c1, W_c2, b_c2, corr_scale, raw_alpha_L, raw_alpha_T,
                       raw_gamma, raw_alpha_R, omega, hpart, bx, (float*)d_out);
}
